// SeqFCEncoder_51127290691717
// MI455X (gfx1250) — matrix operand order repaired and checked by reading, not compiled, not run on hardware
//
#include <hip/hip_runtime.h>

// SeqFCEncoder fused bf16-WMMA kernel for gfx1250 (MI455X).
// out = relu(dropout(X) @ W1 + b1) @ W2 + b2
//  - h (32x1024 bf16) lives in LDS, never touches HBM
//  - weights packed once into CDNA5 WMMA B-fragment layout in d_ws (640 KB)
//  - X tile staged via Tensor Data Mover (zero-pad via TDM OOB semantics)
//  - output staged through LDS for fully coalesced b128 stores (contiguous
//    32 KB span per block) -- output write is the HBM-roofline term.

#define L_SEQ 50
#define H_DIM 1024
#define E_DIM 256
#define MT    32      // batch rows per workgroup

typedef __attribute__((ext_vector_type(8)))  float    v8f;
typedef __attribute__((ext_vector_type(4)))  float    v4f;
typedef __attribute__((ext_vector_type(8)))  __bf16   v8bf;
typedef __attribute__((ext_vector_type(16))) __bf16   v16bf;
typedef __attribute__((ext_vector_type(4)))  unsigned v4u;
typedef __attribute__((ext_vector_type(8)))  int      v8i;
typedef __attribute__((ext_vector_type(4)))  int      v4i;

static __device__ __forceinline__ __bf16 f2bf(float f) {
  unsigned u = __builtin_bit_cast(unsigned, f);
  unsigned r = u + 0x7FFFu + ((u >> 16) & 1u);   // round-to-nearest-even
  return __builtin_bit_cast(__bf16, (unsigned short)(r >> 16));
}

static __device__ __forceinline__ unsigned hash32(unsigned x) {
  x ^= 42u * 0x9E3779B9u;            // fold in dropout key
  x ^= x >> 16; x *= 0x7feb352dU;
  x ^= x >> 15; x *= 0x846ca68bU;
  x ^= x >> 16;
  return x;
}

// Build a v16bf A-fragment from two contiguous 16-byte LDS chunks
// (CDNA5 16-bit A layout: lane holds k in [kh*8, kh*8+8) and [16+kh*8, ...)).
static __device__ __forceinline__ v16bf lds_frag(const __bf16* p0) {
  v8bf lo = *(const v8bf*)(p0);
  v8bf hi = *(const v8bf*)(p0 + 16);
  return __builtin_shufflevector(lo, hi, 0,1,2,3,4,5,6,7,8,9,10,11,12,13,14,15);
}

// ---------------------------------------------------------------------------
// Prep: pack W1 (50x1024, K padded to 64) and W2 (1024x256) into bf16
// B-operand fragment order: element ((nt*KT+kt)*32 + lane)*16 + e holds
// W[k = kt*32 + (lane/16)*8 + e + (e&8)][n = nt*16 + lane%16].
// ---------------------------------------------------------------------------
__global__ __launch_bounds__(256) void pack_weights(
    const float* __restrict__ W1, const float* __restrict__ W2,
    __bf16* __restrict__ W1p, __bf16* __restrict__ W2p)
{
  int tid = blockIdx.x * 256 + threadIdx.x;
  if (tid < 4096) {                          // W1: 64 ntiles * 2 ktiles * 32 lanes
    int lane = tid & 31, tile = tid >> 5;
    int kt = tile & 1, nt = tile >> 1;
    int n = nt * 16 + (lane & 15);
    int kbase = kt * 32 + (lane >> 4) * 8;
#pragma unroll
    for (int e = 0; e < 16; ++e) {
      int k = kbase + e + (e & 8);
      float v = (k < L_SEQ) ? W1[k * H_DIM + n] : 0.0f;
      W1p[tid * 16 + e] = f2bf(v);
    }
  } else if (tid < 4096 + 16384) {           // W2: 16 ntiles * 32 ktiles * 32 lanes
    int t2 = tid - 4096;
    int lane = t2 & 31, tile = t2 >> 5;
    int kt = tile & 31, nt = tile >> 5;
    int n = nt * 16 + (lane & 15);
    int kbase = kt * 32 + (lane >> 4) * 8;
#pragma unroll
    for (int e = 0; e < 16; ++e)
      W2p[t2 * 16 + e] = f2bf(W2[(kbase + e + (e & 8)) * E_DIM + n]);
  }
}

// ---------------------------------------------------------------------------
// Main fused kernel: one block = 32 batch rows, 256 threads (8 wave32s).
// ---------------------------------------------------------------------------
__global__ __launch_bounds__(256) void seqfc_main(
    const float* __restrict__ seq,
    const float* __restrict__ b1, const float* __restrict__ b2,
    const __bf16* __restrict__ W1p, const __bf16* __restrict__ W2p,
    float* __restrict__ out)
{
  extern __shared__ char smem[];
  float*  xstage   = (float*)smem;                                   // [MT][64] f32
  int*    firstneg = (int*)(smem + MT * 64 * 4);                     // [MT]
  __bf16* xbf      = (__bf16*)(smem + MT * 64 * 4 + MT * 4);         // [MT][64] bf16
  __bf16* hbuf     = (__bf16*)(smem + MT * 64 * 4 + MT * 4 + MT * 64 * 2); // [MT][1024]

  const int tid   = threadIdx.x;
  const int wave  = tid >> 5;
  const int lane  = tid & 31;
  const int khalf = lane >> 4;     // K half-group within wave
  const int nlane = lane & 15;     // row (A) / col (B,C) within tile
  const int m0    = blockIdx.x * MT;

  // L2 warm hint for this wave's packed-weight stream (global_prefetch_b8)
  __builtin_prefetch(&W2p[(size_t)((wave * 2 * 32) * 32 + lane) * 16], 0, 0);

  // ---- stage 1: X tile -> LDS via Tensor Data Mover --------------------
  // 2D tensor: dim0 = 50 (stride 50), dim1 = B rows. Tile = 64 x 32 with
  // tile_dim0 = 64 > tensor_dim0: TDM OOB reads return zero, giving the
  // K-pad (cols 50..63 = 0) for free, landing as [32][64] f32 in LDS.
#if __has_builtin(__builtin_amdgcn_tensor_load_to_lds)
  if (wave == 0) {
    unsigned long long ga =
        (unsigned long long)(const void*)(seq + (size_t)m0 * L_SEQ);
    unsigned lds_off = (unsigned)(unsigned long long)(void*)xstage; // LDS byte offset
    v4u g0;
    g0[0] = 1u;                                          // count=1, user D#
    g0[1] = lds_off;                                     // lds_addr
    g0[2] = (unsigned)(ga & 0xffffffffu);                // global_addr[31:0]
    g0[3] = (unsigned)((ga >> 32) & 0x1ffffffu)          // global_addr[56:32]
          | (2u << 30);                                  // type=2 ("image")
    v8i g1;
    g1[0] = (int)(2u << 16);                             // wg_mask=0, data_size=4B
    g1[1] = (int)((unsigned)L_SEQ << 16);                // tensor_dim0[15:0]
    g1[2] = 0;                                           // dim0 hi / dim1 lo
    g1[3] = (int)(4u | (64u << 16));                     // dim1 hi (262144), tile_dim0=64
    g1[4] = MT;                                          // tile_dim1=32
    g1[5] = L_SEQ;                                       // tensor_dim0_stride=50
    g1[6] = 0;
    g1[7] = 0;
    v4i gz4 = {0, 0, 0, 0};
    v8i gz8 = {0, 0, 0, 0, 0, 0, 0, 0};
    // 6-arg variant (clang-23 / therock-10.0 headers): groups 0..3 + extra + cpol
    __builtin_amdgcn_tensor_load_to_lds(g0, g1, gz4, gz4, gz8, 0);
    __builtin_amdgcn_s_wait_tensorcnt((short)0);
  }
#else
  for (int i = tid; i < MT * 64; i += 256) {
    int r = i >> 6, c = i & 63;
    xstage[i] = (c < L_SEQ) ? seq[(size_t)(m0 + r) * L_SEQ + c] : 0.0f;
  }
#endif
  __syncthreads();

  // ---- stage 2: first -1.0 per row ----
  if (tid < MT) {
    int fn = L_SEQ;
    for (int c = 0; c < L_SEQ; ++c)
      if (xstage[tid * 64 + c] == -1.0f) { fn = c; break; }
    firstneg[tid] = fn;
  }
  __syncthreads();

  // ---- stage 3: dropout (hash-based) + bf16 convert ----
  const unsigned TH = (unsigned)(0.2 * 4294967296.0);
  for (int i = tid; i < MT * 64; i += 256) {
    int r = i >> 6, c = i & 63;
    float v = xstage[i];
    if (c < firstneg[r]) {            // valid positions only; pads keep -1.0
      if (hash32((unsigned)((m0 + r) * L_SEQ + c)) < TH) v = 0.0f;
    }
    xbf[i] = f2bf(v);
  }
  __syncthreads();

  // ---- stage 4: GEMM1  h = relu(Xbf(32x64) @ W1(64x1024) + b1) -> LDS bf16 ----
  v16bf afr[2][2];                    // [mtile][ktile] A-fragments, reused across N
#pragma unroll
  for (int mt = 0; mt < 2; ++mt)
#pragma unroll
    for (int kt = 0; kt < 2; ++kt) {
      const __bf16* base = &xbf[(mt * 16 + nlane) * 64 + kt * 32 + khalf * 8];
      afr[mt][kt] = lds_frag(base);
    }

  for (int j = 0; j < 8; ++j) {       // wave handles 8 of 64 N-tiles
    int nt = wave * 8 + j;
    const __bf16* bptr = &W1p[(size_t)((nt * 2 + 0) * 32 + lane) * 16];
    v16bf bfr0 = *(const v16bf*)(bptr);
    v16bf bfr1 = *(const v16bf*)(bptr + 32 * 16);
    float bias = b1[nt * 16 + nlane];
#pragma unroll
    for (int mt = 0; mt < 2; ++mt) {
      v8f acc;
#pragma unroll
      for (int r = 0; r < 8; ++r) acc[r] = bias;
      acc = __builtin_amdgcn_wmma_f32_16x16x32_bf16(false, afr[mt][0], false, bfr0,
                                                    (short)0, acc, false, false);
      acc = __builtin_amdgcn_wmma_f32_16x16x32_bf16(false, afr[mt][1], false, bfr1,
                                                    (short)0, acc, false, false);
      int mrow = mt * 16 + khalf * 8;
      int ncol = nt * 16 + nlane;
#pragma unroll
      for (int r = 0; r < 8; ++r)     // C layout: VGPR r -> M = khalf*8 + r
        hbuf[(mrow + r) * H_DIM + ncol] = f2bf(fmaxf(acc[r], 0.0f));
    }
  }
  __syncthreads();

  // ---- stage 5: GEMM2  out = h(32x1024) @ W2(1024x256) + b2 ----
  v8f acc2[2][2];                     // [ntile-of-wave][mtile]
#pragma unroll
  for (int j2 = 0; j2 < 2; ++j2) {
    float bias = b2[(wave * 2 + j2) * 16 + nlane];
#pragma unroll
    for (int mt = 0; mt < 2; ++mt)
#pragma unroll
      for (int r = 0; r < 8; ++r) acc2[j2][mt][r] = bias;
  }

  for (int kt = 0; kt < 32; ++kt) {
    const __bf16* base0 = &hbuf[nlane * H_DIM + kt * 32 + khalf * 8];
    v16bf a0 = lds_frag(base0);                 // rows 0..15
    v16bf a1 = lds_frag(base0 + 16 * H_DIM);    // rows 16..31
#pragma unroll
    for (int j2 = 0; j2 < 2; ++j2) {
      int nt2 = wave * 2 + j2;
      const __bf16* bptr = &W2p[(size_t)((nt2 * 32 + kt) * 32 + lane) * 16];
      v16bf bfr = *(const v16bf*)bptr;
      acc2[j2][0] = __builtin_amdgcn_wmma_f32_16x16x32_bf16(false, a0, false, bfr,
                                                            (short)0, acc2[j2][0], false, false);
      acc2[j2][1] = __builtin_amdgcn_wmma_f32_16x16x32_bf16(false, a1, false, bfr,
                                                            (short)0, acc2[j2][1], false, false);
    }
  }
  __syncthreads();   // everyone finished reading hbuf; LDS can be reused

  // ---- stage 6: C tiles -> LDS, then fully coalesced b128 stores ----
  float* obuf = (float*)smem;          // [MT][E_DIM] f32 = 32 KB, overlaps dead bufs
#pragma unroll
  for (int j2 = 0; j2 < 2; ++j2) {
    int ncol = (wave * 2 + j2) * 16 + nlane;
#pragma unroll
    for (int mt = 0; mt < 2; ++mt) {
      int mrow = mt * 16 + khalf * 8;
#pragma unroll
      for (int r = 0; r < 8; ++r)
        obuf[(mrow + r) * E_DIM + ncol] = acc2[j2][mt][r];
    }
  }
  __syncthreads();

  // block output = one contiguous 32 KB span of `out`
  const v4f* ob4 = (const v4f*)obuf;
  v4f* out4 = (v4f*)(out + (size_t)m0 * E_DIM);
  for (int i = tid; i < MT * E_DIM / 4; i += 256)
    out4[i] = ob4[i];
}

extern "C" void kernel_launch(void* const* d_in, const int* in_sizes, int n_in,
                              void* d_out, int out_size, void* d_ws, size_t ws_size,
                              hipStream_t stream) {
  const float* seq = (const float*)d_in[0];   // (B,1,50) f32
  const float* W1  = (const float*)d_in[1];   // (50,1024)
  const float* b1  = (const float*)d_in[2];   // (1024,)
  const float* W2  = (const float*)d_in[3];   // (1024,256)
  const float* b2  = (const float*)d_in[4];   // (256,)
  float* out = (float*)d_out;

  __bf16* W1p = (__bf16*)d_ws;                // 64*1024 bf16 = 128 KB
  __bf16* W2p = W1p + 64 * H_DIM;             // 1024*256 bf16 = 512 KB

  pack_weights<<<80, 256, 0, stream>>>(W1, W2, W1p, W2p);

  const int nrows = in_sizes[0] / L_SEQ;      // B = 262144
  const size_t smem = (size_t)MT * 64 * 4 + MT * 4 + (size_t)MT * 64 * 2
                    + (size_t)MT * H_DIM * 2; // 77,952 bytes
  seqfc_main<<<nrows / MT, 256, smem, stream>>>(seq, b1, b2, W1p, W2p, out);
}
